// SelfAttention_70755291234855
// MI455X (gfx1250) — compile-verified, matrix operand order checked by reading, not run on hardware
//
#include <hip/hip_runtime.h>
#include <hip/hip_bf16.h>
#include <stdint.h>

// ---------------------------------------------------------------------------
// Types for CDNA5 WMMA (wave32): v_wmma_f32_16x16x32_bf16
// ---------------------------------------------------------------------------
typedef __attribute__((ext_vector_type(16))) __bf16 bf16x16;
typedef __attribute__((ext_vector_type(2)))  __bf16 bf16x2;
typedef __attribute__((ext_vector_type(8)))  float  f32x8;
typedef __attribute__((ext_vector_type(4)))  int    i32x4;

#define BM 128
#define BN 128
#define BKG 64               // K-tile for generic GEMM: two K=32 WMMA steps
#define LDST (BKG + 8)       // ushort stride (72): keeps 16B row alignment, de-conflicts banks

// Async global->LDS copy path (gfx1250 GLOBAL_LOAD_ASYNC_TO_LDS_B128, ASYNCcnt)
#if defined(__gfx1250__) && __has_builtin(__builtin_amdgcn_global_load_async_to_lds_b128)
#define ASYNC_LDS 1
#else
#define ASYNC_LDS 0
#endif

__device__ __forceinline__ void async_wait0() {
#if __has_builtin(__builtin_amdgcn_s_wait_asynccnt)
    __builtin_amdgcn_s_wait_asynccnt(0);
#else
    asm volatile("s_wait_asynccnt 0" ::: "memory");
#endif
}

__device__ __forceinline__ void async_copy16(const void* g, void* l) {
#if ASYNC_LDS
    // Signature (from clang diagnostic): (AS1 i32x4*, AS3 i32x4*, imm offset, imm cpol)
    __builtin_amdgcn_global_load_async_to_lds_b128(
        (__attribute__((address_space(1))) i32x4*)(uintptr_t)g,
        (__attribute__((address_space(3))) i32x4*)(uintptr_t)l, 0, 0);
#endif
}

// Native bf16 conversion: lets the compiler emit v_cvt_(pk_)bf16_f32
__device__ __forceinline__ uint16_t f32_to_bf16(float f) {
    __bf16 h = (__bf16)f;
    return __builtin_bit_cast(uint16_t, h);
}
__device__ __forceinline__ uint32_t pack_bf16x2(float a, float b) {
    bf16x2 h;
    h[0] = (__bf16)a;
    h[1] = (__bf16)b;
    return __builtin_bit_cast(uint32_t, h);
}
__device__ __forceinline__ float bf16_to_f32(uint16_t h) {
    uint32_t u = ((uint32_t)h) << 16;
    return __builtin_bit_cast(float, u);
}

// ---- global -> LDS tile fills (128 rows x BKG cols, bf16 in LDS) ----------
// One base pointer per tensor; all chunk offsets are compile-time constants
// (LD is a template arg) so they fold into the instruction ioffset fields.
// f32 source: load float4, convert to packed bf16, DS-store.
template <int LD>
__device__ __forceinline__ bool fill_lds(const float* src, int row0, int k0,
                                         uint16_t* lds) {
    int t  = threadIdx.x;
    int r0 = t >> 4;                 // 0..15
    int c0 = (t & 15) * 4;           // 0..60
    const float* p = src + (size_t)(row0 + r0) * LD + k0 + c0;
    uint16_t*    l = lds + r0 * LDST + c0;
#pragma unroll
    for (int i = 0; i < 8; ++i) {    // rows r0, r0+16, ..., r0+112
        float4 v = *(const float4*)(p + (size_t)i * 16 * LD);
        uint2 q;
        q.x = pack_bf16x2(v.x, v.y);
        q.y = pack_bf16x2(v.z, v.w);
        *(uint2*)(l + i * 16 * LDST) = q;
    }
    return false;
}
// bf16 source: straight byte copy -> async global->LDS when available.
template <int LD>
__device__ __forceinline__ bool fill_lds(const uint16_t* src, int row0, int k0,
                                         uint16_t* lds) {
    int t = threadIdx.x;
#if ASYNC_LDS
    int r0 = t >> 3;                 // 0..31
    int k8 = (t & 7) * 8;            // 0..56 (16B chunks)
    const uint16_t* p = src + (size_t)(row0 + r0) * LD + k0 + k8;
    uint16_t*       l = lds + r0 * LDST + k8;
#pragma unroll
    for (int i = 0; i < 4; ++i)      // rows r0, r0+32, r0+64, r0+96
        async_copy16(p + (size_t)i * 32 * LD, l + i * 32 * LDST);
    return true;
#else
    int r0 = t >> 4;
    int c0 = (t & 15) * 4;
    const uint16_t* p = src + (size_t)(row0 + r0) * LD + k0 + c0;
    uint16_t*       l = lds + r0 * LDST + c0;
#pragma unroll
    for (int i = 0; i < 8; ++i)
        *(uint2*)(l + i * 16 * LDST) = *(const uint2*)(p + (size_t)i * 16 * LD);
    return false;
#endif
}

// ---- LDS -> 16x32 bf16 fragment (ISA 16-bit A/B layout) --------------------
// lane<16: row=lane,    K in {kb..kb+7, kb+16..kb+23} with kb=0
// lane>=16: row=lane-16, same pattern with kb=8
__device__ __forceinline__ bf16x16 load_frag(const uint16_t* lds, int row, int ksub, int stride) {
    int lane = threadIdx.x & 31;
    const uint16_t* p = lds + (row + (lane & 15)) * stride + ksub + ((lane >> 4) << 3);
    union { bf16x16 v; uint4 u[2]; } f;
    f.u[0] = *(const uint4*)(p);
    f.u[1] = *(const uint4*)(p + 16);
    return f.v;
}

__device__ __forceinline__ void store_out(float* C, size_t i, float v)    { C[i] = v; }
__device__ __forceinline__ void store_out(uint16_t* C, size_t i, float v) { C[i] = f32_to_bf16(v); }

// ---------------------------------------------------------------------------
// Generic block GEMM: C[m][n] = sum_k A[m][k] * B[n][k]   (i.e. C = A @ B.T)
// 256 threads = 8 waves; wave grid 4(M)x2(N); each wave 32x64 output tile.
// LDA/LDB/LDC are compile-time so all addressing folds into const offsets.
// ---------------------------------------------------------------------------
template <int LDA, int LDB, int LDC, typename TA, typename TB, typename TOUT>
__device__ __forceinline__ void gemm_block(const TA* A, const TB* B, TOUT* C, int K,
                                           int blockM, int blockN) {
    __shared__ uint16_t ldsA[BM * LDST];
    __shared__ uint16_t ldsB[BN * LDST];

    int wave = threadIdx.x >> 5;
    int wm = (wave & 3) * 32;
    int wn = (wave >> 2) * 64;

    f32x8 acc[2][4];
#pragma unroll
    for (int mt = 0; mt < 2; ++mt)
#pragma unroll
        for (int nt = 0; nt < 4; ++nt)
#pragma unroll
            for (int r = 0; r < 8; ++r) acc[mt][nt][r] = 0.0f;

    for (int k0 = 0; k0 < K; k0 += BKG) {
        __syncthreads();
        bool asyncA = fill_lds<LDA>(A, blockM, k0, ldsA);
        bool asyncB = fill_lds<LDB>(B, blockN, k0, ldsB);
        if (asyncA || asyncB) async_wait0();
        __syncthreads();
#pragma unroll
        for (int ks = 0; ks < BKG / 32; ++ks) {
            bf16x16 af[2], bfm[4];
            af[0] = load_frag(ldsA, wm,      ks * 32, LDST);
            af[1] = load_frag(ldsA, wm + 16, ks * 32, LDST);
#pragma unroll
            for (int nt = 0; nt < 4; ++nt)
                bfm[nt] = load_frag(ldsB, wn + nt * 16, ks * 32, LDST);
#pragma unroll
            for (int mt = 0; mt < 2; ++mt)
#pragma unroll
                for (int nt = 0; nt < 4; ++nt)
                    acc[mt][nt] = __builtin_amdgcn_wmma_f32_16x16x32_bf16(
                        false, af[mt], false, bfm[nt], (short)0, acc[mt][nt], false, false);
        }
    }

    // C 16x16 f32 layout: VGPR r, lane<16 -> (M=r, N=lane); lane>=16 -> (M=r+8, N=lane-16)
    int lane = threadIdx.x & 31;
    int coln = lane & 15;
    int rofs = (lane >> 4) << 3;
#pragma unroll
    for (int mt = 0; mt < 2; ++mt)
#pragma unroll
        for (int nt = 0; nt < 4; ++nt) {
            TOUT* cp = C + (size_t)(blockM + wm + mt * 16 + rofs) * LDC
                         + (blockN + wn + nt * 16 + coln);
#pragma unroll
            for (int r = 0; r < 8; ++r)
                store_out(cp, (size_t)r * LDC, acc[mt][nt][r]);
        }
}

// ---------------------------------------------------------------------------
// K1: Q/K/V projection  (x f32 2048x4096) @ (w f32 4096x4096).T -> bf16
// ---------------------------------------------------------------------------
__global__ void qkv_gemm(const float* __restrict__ x, const float* __restrict__ w,
                         uint16_t* __restrict__ out) {
    gemm_block<4096, 4096, 4096>(x, w, out, 4096, blockIdx.y * BM, blockIdx.x * BN);
}

// ---------------------------------------------------------------------------
// K2: RoPE on K and V (reference ropes k and v, not q). In-place on ws bufs.
// ---------------------------------------------------------------------------
__global__ void rope_kernel(uint16_t* __restrict__ kbf, uint16_t* __restrict__ vbf,
                            const float* __restrict__ freqs) {
    size_t idx = (size_t)blockIdx.x * 256 + threadIdx.x;   // 2 * 4M pairs
    uint16_t* buf = (idx < (1u << 22)) ? kbf : vbf;
    size_t p = idx & ((1u << 22) - 1);
    int m  = (int)(p >> 11);         // row (b*1024+s)
    int pp = (int)(p & 2047);        // pair within row
    int s  = m & 1023;
    int fp = pp & 63;                // pair index within head
    float c  = freqs[s * 128 + fp * 2 + 0];
    float sn = freqs[s * 128 + fp * 2 + 1];
    size_t base = (size_t)m * 4096 + (size_t)pp * 2;
    float x0 = bf16_to_f32(buf[base]);
    float x1 = bf16_to_f32(buf[base + 1]);
    buf[base]     = f32_to_bf16(x0 * c - x1 * sn);
    buf[base + 1] = f32_to_bf16(x0 * sn + x1 * c);
}

// ---------------------------------------------------------------------------
// K3: per (b,h)  KtV = K_h^T @ V_h  (128x128), scaled by 1/sqrt(128),
//     stored TRANSPOSED (ktvT[d2*128+d1]) so K4 consumes it as A@B.T.
//     No softmax in the reference => (QK^T)V == Q(K^T V).
// ---------------------------------------------------------------------------
#define BK3 32
#define LDST3 40
__global__ void ktv_kernel(const uint16_t* __restrict__ kbf,
                           const uint16_t* __restrict__ vbf,
                           uint16_t* __restrict__ ktvT) {
    int bh = blockIdx.x;             // b*32 + h
    int b = bh >> 5, h = bh & 31;
    __shared__ uint16_t ldsA[128 * LDST3];   // [d1][s]
    __shared__ uint16_t ldsB[128 * LDST3];   // [d2][s]

    int wave = threadIdx.x >> 5;
    int wm = (wave & 3) * 32;
    int wn = (wave >> 2) * 64;

    f32x8 acc[2][4];
#pragma unroll
    for (int mt = 0; mt < 2; ++mt)
#pragma unroll
        for (int nt = 0; nt < 4; ++nt)
#pragma unroll
            for (int r = 0; r < 8; ++r) acc[mt][nt][r] = 0.0f;

    int so0 = threadIdx.x >> 5;      // 0..7
    int d4  = (threadIdx.x & 31) * 4;  // 0..124
    const uint16_t* kp = kbf + (size_t)b * 1024 * 4096 + h * 128 + (size_t)so0 * 4096 + d4;
    const uint16_t* vp = vbf + (size_t)b * 1024 * 4096 + h * 128 + (size_t)so0 * 4096 + d4;

    for (int s0 = 0; s0 < 1024; s0 += BK3) {
        __syncthreads();
        // transpose-fill: read 4 contiguous d (bf16x4), scatter into [d][s]
        const uint16_t* kq = kp + (size_t)s0 * 4096;
        const uint16_t* vq = vp + (size_t)s0 * 4096;
#pragma unroll
        for (int i = 0; i < 4; ++i) {            // s rows so0, so0+8, so0+16, so0+24
            int so = so0 + i * 8;
            uint2 pk = *(const uint2*)(kq + (size_t)i * 8 * 4096);
            uint2 pv = *(const uint2*)(vq + (size_t)i * 8 * 4096);
            uint16_t ke[4] = {(uint16_t)pk.x, (uint16_t)(pk.x >> 16),
                              (uint16_t)pk.y, (uint16_t)(pk.y >> 16)};
            uint16_t ve[4] = {(uint16_t)pv.x, (uint16_t)(pv.x >> 16),
                              (uint16_t)pv.y, (uint16_t)(pv.y >> 16)};
#pragma unroll
            for (int j = 0; j < 4; ++j) {
                ldsA[(d4 + j) * LDST3 + so] = ke[j];
                ldsB[(d4 + j) * LDST3 + so] = ve[j];
            }
        }
        __syncthreads();
        bf16x16 af[2], bfm[4];
        af[0] = load_frag(ldsA, wm,      0, LDST3);
        af[1] = load_frag(ldsA, wm + 16, 0, LDST3);
#pragma unroll
        for (int nt = 0; nt < 4; ++nt)
            bfm[nt] = load_frag(ldsB, wn + nt * 16, 0, LDST3);
#pragma unroll
        for (int mt = 0; mt < 2; ++mt)
#pragma unroll
            for (int nt = 0; nt < 4; ++nt)
                acc[mt][nt] = __builtin_amdgcn_wmma_f32_16x16x32_bf16(
                    false, af[mt], false, bfm[nt], (short)0, acc[mt][nt], false, false);
    }

    const float scale = 0.08838834764831845f;   // 1/sqrt(128)
    int lane = threadIdx.x & 31;
    int coln = lane & 15;
    int rofs = (lane >> 4) << 3;
#pragma unroll
    for (int mt = 0; mt < 2; ++mt)
#pragma unroll
        for (int nt = 0; nt < 4; ++nt) {
            int m0 = wm + mt * 16 + rofs;
            int n  = wn + nt * 16 + coln;
            uint16_t* cp = ktvT + (size_t)bh * 16384 + (size_t)n * 128 + m0;
#pragma unroll
            for (int r = 0; r < 8; ++r)
                cp[r] = f32_to_bf16(acc[mt][nt][r] * scale);
        }
}

// ---------------------------------------------------------------------------
// K4: attn[b,s,h,:] = Q_h(1024x128) @ KtV_h(128x128)   (K=128)
// ---------------------------------------------------------------------------
__global__ void attn_gemm(const uint16_t* __restrict__ qbf,
                          const uint16_t* __restrict__ ktvT,
                          uint16_t* __restrict__ attn) {
    int bh = blockIdx.z;
    int b = bh >> 5, h = bh & 31;
    const uint16_t* A = qbf  + (size_t)b * 1024 * 4096 + h * 128;
    const uint16_t* B = ktvT + (size_t)bh * 16384;
    uint16_t*       C = attn + (size_t)b * 1024 * 4096 + h * 128;
    gemm_block<4096, 128, 4096>(A, B, C, 128, blockIdx.y * BM, blockIdx.x * BN);
}

// ---------------------------------------------------------------------------
// K5: out(f32) = attn(bf16 2048x4096) @ wo(f32 4096x4096).T
// ---------------------------------------------------------------------------
__global__ void out_gemm(const uint16_t* __restrict__ attn,
                         const float* __restrict__ wo,
                         float* __restrict__ out) {
    gemm_block<4096, 4096, 4096>(attn, wo, out, 4096, blockIdx.y * BM, blockIdx.x * BN);
}

// ---------------------------------------------------------------------------
extern "C" void kernel_launch(void* const* d_in, const int* in_sizes, int n_in,
                              void* d_out, int out_size, void* d_ws, size_t ws_size,
                              hipStream_t stream) {
    const float* x     = (const float*)d_in[0];
    const float* wq    = (const float*)d_in[1];
    const float* wk    = (const float*)d_in[2];
    const float* wv    = (const float*)d_in[3];
    const float* wo    = (const float*)d_in[4];
    const float* freqs = (const float*)d_in[7];   // cache_k/cache_v (5,6) unused: start_pos=0
    float* out = (float*)d_out;

    char* ws = (char*)d_ws;
    uint16_t* qbf  = (uint16_t*)(ws);                             // 16 MB
    uint16_t* kbf  = (uint16_t*)(ws + (size_t)16 * 1024 * 1024);  // 16 MB
    uint16_t* vbf  = (uint16_t*)(ws + (size_t)32 * 1024 * 1024);  // 16 MB
    uint16_t* attn = (uint16_t*)(ws + (size_t)48 * 1024 * 1024);  // 16 MB
    uint16_t* ktvT = (uint16_t*)(ws + (size_t)64 * 1024 * 1024);  // 2 MB

    dim3 blk(256);
    dim3 gproj(4096 / BN, 2048 / BM);   // 32 x 16

    qkv_gemm<<<gproj, blk, 0, stream>>>(x, wq, qbf);
    qkv_gemm<<<gproj, blk, 0, stream>>>(x, wk, kbf);
    qkv_gemm<<<gproj, blk, 0, stream>>>(x, wv, vbf);

    rope_kernel<<<(2u << 22) / 256, blk, 0, stream>>>(kbf, vbf, freqs);

    ktv_kernel<<<64, blk, 0, stream>>>(kbf, vbf, ktvT);

    attn_gemm<<<dim3(1, 1024 / BM, 64), blk, 0, stream>>>(qbf, ktvT, attn);

    out_gemm<<<gproj, blk, 0, stream>>>(attn, wo, out);
}
